// AttentionWithRelPos_60730837565694
// MI455X (gfx1250) — compile-verified, matrix operand order checked by reading, not run on hardware
//
#include <hip/hip_runtime.h>
#include <hip/hip_bf16.h>

// ---------------------------------------------------------------------------
// Types for gfx1250 WMMA
// ---------------------------------------------------------------------------
typedef __attribute__((ext_vector_type(16))) __bf16       v16bf;
typedef __attribute__((ext_vector_type(8)))  float        v8f;
typedef __attribute__((ext_vector_type(4)))  unsigned int v4u;

#define Bq   64
#define Nn   197
#define Cc   768
#define Hh   12
#define HD   64
#define SCALE 0.125f        // 64^-0.5
#define QPAD 208            // keys/queries padded to 13*16
#define KPAD 224            // keys padded to 7*32 for P@V k-chunks
#define MROWS (Bq * Nn)     // 12608
#define MPAD  12800         // 50 * 256 (block-row granularity)
#define MB    50            // row blocks of 256

__device__ __forceinline__ unsigned short f2bf(float f) {
    unsigned int u = __builtin_bit_cast(unsigned int, f);
    u += 0x7FFFu + ((u >> 16) & 1u);   // round-to-nearest-even
    return (unsigned short)(u >> 16);
}

__device__ __forceinline__ v8f zero8() {
    v8f z = {0.f,0.f,0.f,0.f,0.f,0.f,0.f,0.f};
    return z;
}

// 16-element bf16 fragment for the ISA layout: per lane K = [8h..8h+7] then
// [16+8h..16+8h+7], each contiguous -> two b128 loads. `p` = row/col + k0.
__device__ __forceinline__ v16bf frag16(const unsigned short* p, int half) {
    union { v4u h[2]; v16bf bf; } u;
    u.h[0] = *(const v4u*)(p + 8 * half);
    u.h[1] = *(const v4u*)(p + 16 + 8 * half);
    return u.bf;
}

#define WMMA_BF16(A, B, C) \
    __builtin_amdgcn_wmma_f32_16x16x32_bf16(false, (A), false, (B), (short)0, (C), false, false)

// ---- CDNA5 async copy: global -> LDS, tracked by ASYNCcnt -----------------
__device__ __forceinline__ void stage_async16(const unsigned short* gsrc,
                                              unsigned short* lds_dst) {
    unsigned lds_off = (unsigned)(unsigned long long)(void*)lds_dst; // low 32b = LDS offset
    unsigned long long ga = (unsigned long long)gsrc;
    asm volatile("global_load_async_to_lds_b128 %0, %1, off"
                 :: "v"(lds_off), "v"(ga) : "memory");
}
__device__ __forceinline__ void wait_async0() {
    asm volatile("s_wait_asynccnt 0x0" ::: "memory");
}

// ---------------------------------------------------------------------------
// Prep kernels
// ---------------------------------------------------------------------------
__global__ void cvt_bf16_kernel(const float* __restrict__ in,
                                unsigned short* __restrict__ out, int n) {
    int i = blockIdx.x * blockDim.x + threadIdx.x;
    int stride = gridDim.x * blockDim.x;
    for (; i < n; i += stride) out[i] = f2bf(in[i]);
}

__global__ void gather_bias_kernel(const float* __restrict__ rel_pos,
                                   const int* __restrict__ idx,
                                   float* __restrict__ out) {
    const int PER_H = 196 * 196;
    int i = blockIdx.x * blockDim.x + threadIdx.x;
    int stride = gridDim.x * blockDim.x;
    int n = Hh * PER_H;
    for (; i < n; i += stride) {
        int h = i / PER_H, p = i - h * PER_H;
        out[i] = rel_pos[h * (27 * 27) + idx[p]];
    }
}

// ---------------------------------------------------------------------------
// QKV GEMM: block = 256 rows x 64 cols. B tile (64x32 bf16 = 4KB) staged into
// LDS via async copy, double-buffered, shared by 8 waves; each wave owns a
// 32x64 tile (8 WMMAs / k-step).
// ---------------------------------------------------------------------------
__global__ __launch_bounds__(256) void qkv_gemm_kernel(
    const unsigned short* __restrict__ xb, const unsigned short* __restrict__ wb,
    const float* __restrict__ qkv_b,
    unsigned short* __restrict__ q, unsigned short* __restrict__ k,
    unsigned short* __restrict__ vt) {
    __shared__ __align__(16) unsigned short bstage[2][64][32];   // 8 KB

    int tid  = threadIdx.x;
    int wv   = tid >> 5;
    int lane = tid & 31;
    int half = lane >> 4, l16 = lane & 15;
    int mb = blockIdx.x % MB;
    int ng = blockIdx.x / MB;                  // 0..35
    int r0 = mb * 256 + wv * 32;
    int c0 = ng * 64;

    // staging assignment: 256 threads x 16B cover 64 cols x 32 k
    int scol = tid >> 2, schk = (tid & 3) * 8;
    const unsigned short* bsrc = wb + (size_t)(c0 + scol) * Cc + schk;

    v8f acc[2][4];
#pragma unroll
    for (int rt = 0; rt < 2; ++rt)
#pragma unroll
        for (int t = 0; t < 4; ++t) acc[rt][t] = zero8();

    const unsigned short* arow0 = xb + (size_t)(r0 + l16) * Cc;
    const unsigned short* arow1 = xb + (size_t)(r0 + 16 + l16) * Cc;

    stage_async16(bsrc, &bstage[0][scol][schk]);
    wait_async0();
    __syncthreads();

    for (int i = 0; i < 24; ++i) {             // 768 / 32 k-steps
        int buf = i & 1;
        if (i < 23)
            stage_async16(bsrc + (i + 1) * 32, &bstage[buf ^ 1][scol][schk]);
        int k0 = i * 32;
        v16bf a0 = frag16(arow0 + k0, half);
        v16bf a1 = frag16(arow1 + k0, half);
#pragma unroll
        for (int t = 0; t < 4; ++t) {
            v16bf b = frag16(&bstage[buf][t * 16 + l16][0], half);
            acc[0][t] = WMMA_BF16(a0, b, acc[0][t]);
            acc[1][t] = WMMA_BF16(a1, b, acc[1][t]);
        }
        if (i < 23) {
            wait_async0();
            __syncthreads();
        }
    }

    int which = c0 / Cc;                 // 0:q 1:k 2:v
    int h = (c0 % Cc) / HD;
    float sc = (which == 0) ? SCALE : 1.0f;
#pragma unroll
    for (int rt = 0; rt < 2; ++rt)
#pragma unroll
        for (int t = 0; t < 4; ++t) {
            int d = t * 16 + l16;
            float cb = qkv_b[c0 + t * 16 + l16];
#pragma unroll
            for (int e = 0; e < 8; ++e) {
                int r = r0 + rt * 16 + e + 8 * half;
                if (r >= MROWS) continue;
                int b_ = r / Nn, n = r - b_ * Nn;
                unsigned short o = f2bf((acc[rt][t][e] + cb) * sc);
                size_t bh = (size_t)(b_ * Hh + h);
                if (which == 2)
                    vt[bh * HD * KPAD + (size_t)d * KPAD + n] = o;
                else if (which == 0)
                    q[bh * QPAD * HD + (size_t)n * HD + d] = o;
                else
                    k[bh * QPAD * HD + (size_t)n * HD + d] = o;
            }
        }
}

// ---------------------------------------------------------------------------
// Attention: one wave per (b,h,query-block-of-16).
// ---------------------------------------------------------------------------
__global__ __launch_bounds__(256) void attn_kernel(
    const unsigned short* __restrict__ q, const unsigned short* __restrict__ k,
    const unsigned short* __restrict__ vt, const float* __restrict__ bias_full,
    unsigned short* __restrict__ aout) {
    __shared__ __align__(16) unsigned short lds_p[8][16][KPAD];  // 57344 B

    int wv   = threadIdx.x >> 5;
    int lane = threadIdx.x & 31;
    int half = lane >> 4, l16 = lane & 15;
    int wid  = blockIdx.x * 8 + wv;
    int qb   = wid % 13;
    int bh   = wid / 13;
    if (bh >= Bq * Hh) return;
    int h = bh % Hh, b_ = bh / Hh;

    const unsigned short* qbase = q  + (size_t)bh * QPAD * HD;
    const unsigned short* kbase = k  + (size_t)bh * QPAD * HD;
    const unsigned short* vbase = vt + (size_t)bh * HD * KPAD;

    v8f S[13];
#pragma unroll
    for (int t = 0; t < 13; ++t) S[t] = zero8();
    const unsigned short* arow = qbase + (size_t)(qb * 16 + l16) * HD;
#pragma unroll
    for (int k0 = 0; k0 < HD; k0 += 32) {
        v16bf a = frag16(arow + k0, half);
#pragma unroll
        for (int t = 0; t < 13; ++t) {
            v16bf b = frag16(kbase + (size_t)(t * 16 + l16) * HD + k0, half);
            S[t] = WMMA_BF16(a, b, S[t]);
        }
    }

    const float* hb = bias_full + (size_t)h * 196 * 196;
    float rmax[8];
#pragma unroll
    for (int e = 0; e < 8; ++e) rmax[e] = -3.0e38f;
#pragma unroll
    for (int t = 0; t < 13; ++t) {
        int nk = t * 16 + l16;
#pragma unroll
        for (int e = 0; e < 8; ++e) {
            int nq = qb * 16 + e + 8 * half;
            float s = S[t][e];
            if (nk >= Nn || nq >= Nn)      s = -1.0e30f;
            else if (nk >= 1 && nq >= 1)   s += hb[(nq - 1) * 196 + (nk - 1)];
            S[t][e] = s;
            rmax[e] = fmaxf(rmax[e], s);
        }
    }
#pragma unroll
    for (int off = 1; off < 16; off <<= 1)
#pragma unroll
        for (int e = 0; e < 8; ++e)
            rmax[e] = fmaxf(rmax[e], __shfl_xor(rmax[e], off, 32));

    float rsum[8];
#pragma unroll
    for (int e = 0; e < 8; ++e) rsum[e] = 0.f;
#pragma unroll
    for (int t = 0; t < 13; ++t)
#pragma unroll
        for (int e = 0; e < 8; ++e) {
            float p = __expf(S[t][e] - rmax[e]);
            S[t][e] = p;
            rsum[e] += p;
        }
#pragma unroll
    for (int off = 1; off < 16; off <<= 1)
#pragma unroll
        for (int e = 0; e < 8; ++e)
            rsum[e] += __shfl_xor(rsum[e], off, 32);
    float rinv[8];
#pragma unroll
    for (int e = 0; e < 8; ++e) rinv[e] = 1.0f / rsum[e];

#pragma unroll
    for (int t = 0; t < 13; ++t)
#pragma unroll
        for (int e = 0; e < 8; ++e)
            lds_p[wv][e + 8 * half][t * 16 + l16] = f2bf(S[t][e] * rinv[e]);
#pragma unroll
    for (int e = 0; e < 8; ++e)
        lds_p[wv][e + 8 * half][QPAD + l16] = 0;

    v8f O[4];
#pragma unroll
    for (int t = 0; t < 4; ++t) O[t] = zero8();
    const unsigned short* prow = &lds_p[wv][l16][0];
#pragma unroll
    for (int kc = 0; kc < 7; ++kc) {
        v16bf a = frag16(prow + kc * 32, half);
#pragma unroll
        for (int t = 0; t < 4; ++t) {
            v16bf b = frag16(vbase + (size_t)(t * 16 + l16) * KPAD + kc * 32, half);
            O[t] = WMMA_BF16(a, b, O[t]);
        }
    }

#pragma unroll
    for (int t = 0; t < 4; ++t) {
        int c = h * HD + t * 16 + l16;
#pragma unroll
        for (int e = 0; e < 8; ++e) {
            int nq = qb * 16 + e + 8 * half;
            if (nq < Nn)
                aout[(size_t)(b_ * Nn + nq) * Cc + c] = f2bf(O[t][e]);
        }
    }
}

// ---------------------------------------------------------------------------
// Proj GEMM: same blocked scheme, f32 output + bias
// ---------------------------------------------------------------------------
__global__ __launch_bounds__(256) void proj_gemm_kernel(
    const unsigned short* __restrict__ ab, const unsigned short* __restrict__ wb,
    const float* __restrict__ pbias, float* __restrict__ out) {
    __shared__ __align__(16) unsigned short bstage[2][64][32];

    int tid  = threadIdx.x;
    int wv   = tid >> 5;
    int lane = tid & 31;
    int half = lane >> 4, l16 = lane & 15;
    int mb = blockIdx.x % MB;
    int ng = blockIdx.x / MB;                  // 0..11
    int r0 = mb * 256 + wv * 32;
    int c0 = ng * 64;

    int scol = tid >> 2, schk = (tid & 3) * 8;
    const unsigned short* bsrc = wb + (size_t)(c0 + scol) * Cc + schk;

    v8f acc[2][4];
#pragma unroll
    for (int rt = 0; rt < 2; ++rt)
#pragma unroll
        for (int t = 0; t < 4; ++t) acc[rt][t] = zero8();

    const unsigned short* arow0 = ab + (size_t)(r0 + l16) * Cc;
    const unsigned short* arow1 = ab + (size_t)(r0 + 16 + l16) * Cc;

    stage_async16(bsrc, &bstage[0][scol][schk]);
    wait_async0();
    __syncthreads();

    for (int i = 0; i < 24; ++i) {
        int buf = i & 1;
        if (i < 23)
            stage_async16(bsrc + (i + 1) * 32, &bstage[buf ^ 1][scol][schk]);
        int k0 = i * 32;
        v16bf a0 = frag16(arow0 + k0, half);
        v16bf a1 = frag16(arow1 + k0, half);
#pragma unroll
        for (int t = 0; t < 4; ++t) {
            v16bf b = frag16(&bstage[buf][t * 16 + l16][0], half);
            acc[0][t] = WMMA_BF16(a0, b, acc[0][t]);
            acc[1][t] = WMMA_BF16(a1, b, acc[1][t]);
        }
        if (i < 23) {
            wait_async0();
            __syncthreads();
        }
    }

#pragma unroll
    for (int rt = 0; rt < 2; ++rt)
#pragma unroll
        for (int t = 0; t < 4; ++t) {
            int c = c0 + t * 16 + l16;
            float cb = pbias[c];
#pragma unroll
            for (int e = 0; e < 8; ++e) {
                int r = r0 + rt * 16 + e + 8 * half;
                if (r < MROWS)
                    out[(size_t)r * Cc + c] = acc[rt][t][e] + cb;
            }
        }
}

// ---------------------------------------------------------------------------
// Host-side launch
// ---------------------------------------------------------------------------
extern "C" void kernel_launch(void* const* d_in, const int* in_sizes, int n_in,
                              void* d_out, int out_size, void* d_ws, size_t ws_size,
                              hipStream_t stream) {
    const float* x       = (const float*)d_in[0];
    const float* qkv_w   = (const float*)d_in[1];
    const float* qkv_b   = (const float*)d_in[2];
    const float* proj_w  = (const float*)d_in[3];
    const float* proj_b  = (const float*)d_in[4];
    const float* rel_pos = (const float*)d_in[5];
    const int*   rp_idx  = (const int*)d_in[6];
    float* out = (float*)d_out;

    char* ws = (char*)d_ws;
    const size_t SZ_XBF   = (size_t)MPAD * Cc * 2;           // 19,660,800 (padded rows)
    const size_t SZ_QKVW  = (size_t)3 * Cc * Cc * 2;         //  3,538,944
    const size_t SZ_PROJW = (size_t)Cc * Cc * 2;             //  1,179,648
    const size_t SZ_BIAS  = (size_t)Hh * 196 * 196 * 4;      //  1,843,968
    const size_t SZ_Q     = (size_t)Bq * Hh * QPAD * HD * 2; // 20,447,232
    const size_t SZ_VT    = (size_t)Bq * Hh * HD * KPAD * 2; // 22,020,096

    unsigned short* xbf   = (unsigned short*)(ws);
    unsigned short* qkvwb = (unsigned short*)(ws + SZ_XBF);
    unsigned short* projwb= (unsigned short*)(ws + SZ_XBF + SZ_QKVW);
    float*          biasf = (float*)        (ws + SZ_XBF + SZ_QKVW + SZ_PROJW);
    char*           qkv0  =                  ws + SZ_XBF + SZ_QKVW + SZ_PROJW + SZ_BIAS;
    unsigned short* qbuf  = (unsigned short*)(qkv0);
    unsigned short* kbuf  = (unsigned short*)(qkv0 + SZ_Q);
    unsigned short* vtbuf = (unsigned short*)(qkv0 + 2 * SZ_Q);
    unsigned short* aout  = (unsigned short*)(qkv0 + 2 * SZ_Q + SZ_VT);
    (void)ws_size; (void)n_in; (void)in_sizes; (void)out_size;

    // zero q/k/vt (covers padded rows/keys)
    hipMemsetAsync(qkv0, 0, 2 * SZ_Q + SZ_VT, stream);

    cvt_bf16_kernel<<<2048, 256, 0, stream>>>(x, xbf, MROWS * Cc);
    cvt_bf16_kernel<<<1024, 256, 0, stream>>>(qkv_w, qkvwb, 3 * Cc * Cc);
    cvt_bf16_kernel<<<512,  256, 0, stream>>>(proj_w, projwb, Cc * Cc);
    gather_bias_kernel<<<512, 256, 0, stream>>>(rel_pos, rp_idx, biasf);

    // QKV GEMM: 50 row-blocks x 36 col-groups
    qkv_gemm_kernel<<<MB * 36, 256, 0, stream>>>(xbf, qkvwb, qkv_b,
                                                 qbuf, kbuf, vtbuf);
    // Attention: 768 * 13 waves / 8 per block
    attn_kernel<<<(Bq * Hh * 13) / 8, 256, 0, stream>>>(qbuf, kbuf, vtbuf, biasf, aout);

    // Proj GEMM: 50 row-blocks x 12 col-groups
    proj_gemm_kernel<<<MB * 12, 256, 0, stream>>>(aout, projwb, proj_b, out);
}
